// Attention_9689446220043
// MI455X (gfx1250) — matrix operand order repaired and checked by reading, not compiled, not run on hardware
//
#include <hip/hip_runtime.h>
#include <hip/hip_bf16.h>
#include <cstdint>

typedef __attribute__((ext_vector_type(16))) __bf16 v16bf;
typedef __attribute__((ext_vector_type(8)))  __bf16 v8bf;
typedef __attribute__((ext_vector_type(8)))  float  v8f;

#define B_ 16
#define A_ 2048
#define D_ 512
#define R_ 10
#define RP 12                    // rank padded to 12 for float4 loads (pads zeroed)
#define MT 32                    // output rows per block (2 WMMA M-fragments)
#define NT 256                   // softmax / GEMM column chunk
#define LOG2E 1.4426950408889634f
#define RSQRT_DK 0.04419417382415922f  // 1/sqrt(512)

// ---------------------------------------------------------------------------
// Kernel 1: P = (x@W1)/sqrt(512)  [B*A, 12],  Q = x@W2^T [B*A, 12],
//           xT = bf16(x) transposed to [b][d][c] for WMMA-B-friendly loads.
// One wave per (b,a) row.
// ---------------------------------------------------------------------------
__global__ __launch_bounds__(256) void k_project(
    const float* __restrict__ x, const float* __restrict__ W1,
    const float* __restrict__ W2, __bf16* __restrict__ xT,
    float* __restrict__ P, float* __restrict__ Q)
{
  __shared__ float W1s[D_ * R_];   // [d][r]
  __shared__ float W2s[R_ * D_];   // [r][d]
  const int t = threadIdx.x;
  for (int i = t; i < D_ * R_; i += 256) { W1s[i] = W1[i]; W2s[i] = W2[i]; }
  __syncthreads();

  const int row  = blockIdx.x * 8 + (t >> 5);   // global row in [0, B*A)
  const int lane = t & 31;
  const int b = row >> 11;
  const int a = row & (A_ - 1);
  const float* xr = x + (size_t)row * D_;

  float p[R_], q[R_];
  #pragma unroll
  for (int r = 0; r < R_; ++r) { p[r] = 0.f; q[r] = 0.f; }

  #pragma unroll
  for (int j = 0; j < 4; ++j) {
    const int dbase = lane * 4 + j * 128;
    const float4 xv = *(const float4*)(xr + dbase);
    const float xe[4] = {xv.x, xv.y, xv.z, xv.w};
    #pragma unroll
    for (int k = 0; k < 4; ++k) {
      const int d = dbase + k;
      const float xd = xe[k];
      #pragma unroll
      for (int r = 0; r < R_; ++r) {
        p[r] = fmaf(xd, W1s[d * R_ + r], p[r]);
        q[r] = fmaf(xd, W2s[r * D_ + d], q[r]);
      }
      xT[(size_t)(b * D_ + d) * A_ + a] = (__bf16)xd;
    }
  }
  #pragma unroll
  for (int off = 16; off >= 1; off >>= 1) {
    #pragma unroll
    for (int r = 0; r < R_; ++r) {
      p[r] += __shfl_xor(p[r], off);
      q[r] += __shfl_xor(q[r], off);
    }
  }
  if (lane == 0) {
    const size_t base = (size_t)row * RP;
    #pragma unroll
    for (int r = 0; r < R_; ++r) { P[base + r] = p[r] * RSQRT_DK; Q[base + r] = q[r]; }
    P[base + 10] = 0.f; P[base + 11] = 0.f;
    Q[base + 10] = 0.f; Q[base + 11] = 0.f;
  }
}

// ---------------------------------------------------------------------------
// Kernel 2: flash-attention over rank-10 scores.
// Block: 256 thr (8 waves) = (batch b, MT=32 output rows a0..a0+31).
// Wave w owns D-columns [w*64, w*64+64) -> 2 M-frags x 4 N-frags of f32 C.
// Per 256-col chunk: score tile (VALU) -> online softmax -> bf16 attn tile in
// LDS (A-fragment layout) -> 8 K-steps x 8 v_wmma_f32_16x16x32_bf16, with
// each B fragment (global, L2-resident) reused by both M-fragments.
// ---------------------------------------------------------------------------
__global__ __launch_bounds__(256) void k_attn(
    const float* __restrict__ P, const float* __restrict__ Q,
    const __bf16* __restrict__ xT, float* __restrict__ out)
{
  __shared__ float  Ps[MT * RP];          // P rows for this tile
  __shared__ float  Sraw[MT][NT + 8];     // raw scores (f32)
  __shared__ __bf16 attnS[MT][NT + 8];    // exp(s-m) (bf16)
  __shared__ float  mrow[MT], lrow[MT], scaleS[MT];

  const int t    = threadIdx.x;
  const int b    = blockIdx.y;
  const int a0   = blockIdx.x * MT;
  const int row0 = b * A_ + a0;

  for (int i = t; i < MT * RP; i += 256) Ps[i] = P[(size_t)row0 * RP + i];
  if (t < MT) { mrow[t] = -INFINITY; lrow[t] = 0.f; }
  __syncthreads();

  const int wv = t >> 5;         // wave id 0..7
  const int ln = t & 15;         // low lane (column index in fragments)
  const int g  = (t >> 4) & 1;   // half-wave select
  const int d0 = wv * 64;

  v8f acc[2][4];                 // [m-frag][n-frag]
  #pragma unroll
  for (int mf = 0; mf < 2; ++mf)
    #pragma unroll
    for (int nf = 0; nf < 4; ++nf) acc[mf][nf] = (v8f)0.f;

  const int ra = t >> 3;         // row 0..31 (phases B/C)
  const int ri = t & 7;          // 8 threads per row, 32 cols each

  for (int it = 0; it < A_ / NT; ++it) {
    const int c0 = it * NT;

    // ---- phase A: raw score tile S[MT][NT], one column per thread ----
    {
      const int c = c0 + t;
      const float* Qr = Q + (size_t)(b * A_ + c) * RP;
      const float4 q0 = *(const float4*)(Qr);
      const float4 q1 = *(const float4*)(Qr + 4);
      const float4 q2 = *(const float4*)(Qr + 8);
      #pragma unroll
      for (int a = 0; a < MT; ++a) {
        const float* pr = &Ps[a * RP];
        const float4 p0 = *(const float4*)(pr);
        const float4 p1 = *(const float4*)(pr + 4);
        const float4 p2 = *(const float4*)(pr + 8);
        float s = p0.x*q0.x + p0.y*q0.y + p0.z*q0.z + p0.w*q0.w
                + p1.x*q1.x + p1.y*q1.y + p1.z*q1.z + p1.w*q1.w
                + p2.x*q2.x + p2.y*q2.y + p2.z*q2.z + p2.w*q2.w;
        Sraw[a][t] = s;
      }
    }
    __syncthreads();

    // ---- phase B: chunk row-max, accumulator scale factors ----
    {
      const float4* src = (const float4*)&Sraw[ra][ri * 32];
      float4 v = src[0];
      float mx = fmaxf(fmaxf(v.x, v.y), fmaxf(v.z, v.w));
      #pragma unroll
      for (int k4 = 1; k4 < 8; ++k4) {
        v = src[k4];
        mx = fmaxf(mx, fmaxf(fmaxf(v.x, v.y), fmaxf(v.z, v.w)));
      }
      #pragma unroll
      for (int off = 4; off >= 1; off >>= 1) mx = fmaxf(mx, __shfl_xor(mx, off));
      if (ri == 0) {
        const float mo = mrow[ra];
        const float mn = fmaxf(mo, mx);
        scaleS[ra] = exp2f((mo - mn) * LOG2E);
        mrow[ra]   = mn;
      }
    }
    __syncthreads();

    // ---- phase C: exp, bf16 attn tile, row-sum update ----
    {
      const float mn = mrow[ra];
      const float4* src = (const float4*)&Sraw[ra][ri * 32];
      float lsum = 0.f;
      #pragma unroll
      for (int k8 = 0; k8 < 4; ++k8) {
        const float4 va = src[k8 * 2];
        const float4 vb = src[k8 * 2 + 1];
        const float e0 = exp2f((va.x - mn) * LOG2E);
        const float e1 = exp2f((va.y - mn) * LOG2E);
        const float e2 = exp2f((va.z - mn) * LOG2E);
        const float e3 = exp2f((va.w - mn) * LOG2E);
        const float e4 = exp2f((vb.x - mn) * LOG2E);
        const float e5 = exp2f((vb.y - mn) * LOG2E);
        const float e6 = exp2f((vb.z - mn) * LOG2E);
        const float e7 = exp2f((vb.w - mn) * LOG2E);
        lsum += e0 + e1 + e2 + e3 + e4 + e5 + e6 + e7;
        v8bf pk;
        pk[0] = (__bf16)e0; pk[1] = (__bf16)e1; pk[2] = (__bf16)e2; pk[3] = (__bf16)e3;
        pk[4] = (__bf16)e4; pk[5] = (__bf16)e5; pk[6] = (__bf16)e6; pk[7] = (__bf16)e7;
        *(v8bf*)&attnS[ra][ri * 32 + k8 * 8] = pk;
      }
      #pragma unroll
      for (int off = 4; off >= 1; off >>= 1) lsum += __shfl_xor(lsum, off);
      if (ri == 0) lrow[ra] = lrow[ra] * scaleS[ra] + lsum;
    }
    __syncthreads();

    // ---- phase D: rescale accumulators, run WMMA over 8 K-steps ----
    {
      #pragma unroll
      for (int v = 0; v < 8; ++v) {
        const float sc0 = scaleS[v + 8 * g];        // C layout: VGPR v = row v+8g
        const float sc1 = scaleS[16 + v + 8 * g];
        #pragma unroll
        for (int nf = 0; nf < 4; ++nf) {
          acc[0][nf][v] *= sc0;
          acc[1][nf][v] *= sc1;
        }
      }
      #pragma unroll
      for (int kk = 0; kk < 8; ++kk) {
        const int cb = kk * 32;
        // A fragments: lane row = mf*16 + ln, element e -> K = 16g + e
        v16bf af0, af1;
        #pragma unroll
        for (int e = 0; e < 16; ++e) {
          const int kcol = cb + 16 * g + e;
          af0[e] = attnS[ln][kcol];
          af1[e] = attnS[16 + ln][kcol];
        }
        #pragma unroll
        for (int nf = 0; nf < 4; ++nf) {
          // B fragment: lane col d = d0+nf*16+ln, 16 consecutive K (g half);
          // reused by both M-fragments -> halves L2 traffic per MAC.
          const size_t boff =
              (size_t)(b * D_ + d0 + nf * 16 + ln) * A_ + (c0 + cb + 16 * g);
          const v16bf bfrag = *(const v16bf*)(xT + boff);
          acc[0][nf] = __builtin_amdgcn_wmma_f32_16x16x32_bf16(
              false, af0, false, bfrag, (short)0, acc[0][nf], false, false);
          acc[1][nf] = __builtin_amdgcn_wmma_f32_16x16x32_bf16(
              false, af1, false, bfrag, (short)0, acc[1][nf], false, false);
        }
      }
    }
    __syncthreads();
  }

  // ---- epilogue: divide by softmax sum, store f32 output ----
  #pragma unroll
  for (int mf = 0; mf < 2; ++mf) {
    #pragma unroll
    for (int v = 0; v < 8; ++v) {
      const int r = mf * 16 + v + 8 * g;
      const int arow = a0 + r;
      const float invl = 1.0f / lrow[r];
      #pragma unroll
      for (int nf = 0; nf < 4; ++nf) {
        out[(size_t)(b * A_ + arow) * D_ + d0 + nf * 16 + ln] = acc[mf][nf][v] * invl;
      }
    }
  }
}

// ---------------------------------------------------------------------------
extern "C" void kernel_launch(void* const* d_in, const int* in_sizes, int n_in,
                              void* d_out, int out_size, void* d_ws, size_t ws_size,
                              hipStream_t stream)
{
  const float* x  = (const float*)d_in[0];   // [16,2048,512]
  const float* W1 = (const float*)d_in[1];   // [512,10]
  const float* W2 = (const float*)d_in[2];   // [10,512]
  float* out = (float*)d_out;                // [16,2048,512]

  char* ws = (char*)d_ws;
  __bf16* xT = (__bf16*)ws;                              // 16*512*2048*2 = 32 MiB
  float*  P  = (float*)(ws + (size_t)B_ * D_ * A_ * 2);  // [B*A, 12]
  float*  Q  = P + (size_t)B_ * A_ * RP;                 // [B*A, 12]

  k_project<<<dim3(B_ * A_ / 8), dim3(256), 0, stream>>>(x, W1, W2, xT, P, Q);
  k_attn<<<dim3(A_ / MT, B_), dim3(256), 0, stream>>>(P, Q, xT, out);
}
